// DeformConv_54331336294806
// MI455X (gfx1250) — matrix operand order repaired and checked by reading, not compiled, not run on hardware
//
#include <hip/hip_runtime.h>

typedef _Float16 h16 __attribute__((ext_vector_type(16)));
typedef _Float16 h8  __attribute__((ext_vector_type(8)));
typedef float    f8  __attribute__((ext_vector_type(8)));

#define NIMG   8
#define CIN    128
#define COUT   256
#define HDIM   96
#define WDIM   96
#define HW     (HDIM * WDIM)        // 9216
#define KTAPS  9
#define CBLK   32                   // K per WMMA step
#define NCB    (CIN / CBLK)         // 4 channel blocks
#define TM     64                   // positions per block
#define A_STR  40                   // LDS row stride (halves), padded for banks
#define B_STR  40

// LDS: A(64*40*2=5120B) + B(256*40*2=20480B) staged per K-step; epilogue
// reuses the same 33280B as a 128x65 f32 transpose buffer (2 passes).
#define SMEM_BYTES 33280

// Pre-converted f16 weights in workspace: [tap][cb][cout(256)][c(32)]
#define WS_NEED ((size_t)(KTAPS * NCB * COUT * CBLK) * sizeof(_Float16))  // 589824 B

// ---------------------------------------------------------------- prep kernel
__global__ __launch_bounds__(256)
void prep_weights_f16(const float* __restrict__ wgt, _Float16* __restrict__ bw)
{
    const int idx = blockIdx.x * 256 + threadIdx.x;   // 0 .. 294911
    const int c   = idx & 31;
    const int n   = (idx >> 5) & 255;
    const int cb  = (idx >> 13) & 3;
    const int tap = idx >> 15;
    bw[idx] = (_Float16)wgt[((size_t)n * CIN + cb * CBLK + c) * KTAPS + tap];
}

// ---------------------------------------------------------------- main kernel
template <bool USE_WS>
__global__ __launch_bounds__(256)
void deform_conv_wmma(const float* __restrict__ inp,
                      const float* __restrict__ off,
                      const float* __restrict__ wgt,
                      const _Float16* __restrict__ bw,
                      float* __restrict__ out)
{
    __shared__ __align__(16) unsigned char smem[SMEM_BYTES];
    _Float16* As = (_Float16*)smem;             // 64  x A_STR halves
    _Float16* Bs = (_Float16*)(smem + 5120);    // 256 x B_STR halves
    float*    Cs = (float*)smem;                // 128 x 65 f32 (epilogue)

    const int tid  = threadIdx.x;
    const int lane = tid & 31;
    const int l16  = lane & 15;
    const int lhi  = lane >> 4;        // 0/1: selects K half per ISA layout
    const int wid  = tid >> 5;         // 0..7
    const int wm   = wid & 3;          // M row of 16
    const int wn   = wid >> 2;         // Cout half (0/1 -> 128 each)

    const int p0    = blockIdx.x * TM;       // global flat position base
    const int img   = p0 / HW;               // tile never crosses images
    const int prem0 = p0 - img * HW;

    // ---- A-fill mapping: thread t -> position fm = t/4, channels (t%4)*8..+7
    const int fm  = tid >> 2;                // 0..63
    const int fc0 = (tid & 3) * 8;
    const int fp  = prem0 + fm;              // position within image
    const int fho = fp / WDIM;
    const int fwo = fp - fho * WDIM;

    const float* inImg  = inp + (size_t)img * CIN * HW;
    const float* offImg = off + (size_t)img * (2 * KTAPS) * HW;

    f8 acc[8] = {};   // 8 x (16x16 f32) accumulator tiles per wave

    #pragma unroll 1
    for (int tap = 0; tap < KTAPS; ++tap) {
        // ---- per-(position,tap) bilinear setup, hoisted over channel blocks
        const int ky = tap / 3;
        const int kx = tap - 3 * ky;
        const float dy = offImg[(2 * tap)     * HW + fp];
        const float dx = offImg[(2 * tap + 1) * HW + fp];
        const float py = (float)(fho - 1 + ky) + dy;
        const float px = (float)(fwo - 1 + kx) + dx;
        const float fy = floorf(py);
        const float fx = floorf(px);
        const float ly = py - fy;
        const float lx = px - fx;
        const int y0 = (int)fy, x0 = (int)fx;
        const int y1 = y0 + 1,  x1 = x0 + 1;
        const float vy0 = (y0 >= 0 && y0 < HDIM) ? 1.0f : 0.0f;
        const float vy1 = (y1 >= 0 && y1 < HDIM) ? 1.0f : 0.0f;
        const float vx0 = (x0 >= 0 && x0 < WDIM) ? 1.0f : 0.0f;
        const float vx1 = (x1 >= 0 && x1 < WDIM) ? 1.0f : 0.0f;
        const float w00 = (1.0f - ly) * (1.0f - lx) * vy0 * vx0;
        const float w01 = (1.0f - ly) * lx          * vy0 * vx1;
        const float w10 = ly * (1.0f - lx)          * vy1 * vx0;
        const float w11 = ly * lx                   * vy1 * vx1;
        const int y0c = min(max(y0, 0), HDIM - 1);
        const int y1c = min(max(y1, 0), HDIM - 1);
        const int x0c = min(max(x0, 0), WDIM - 1);
        const int x1c = min(max(x1, 0), WDIM - 1);
        const int o00 = y0c * WDIM + x0c;
        const int o01 = y0c * WDIM + x1c;
        const int o10 = y1c * WDIM + x0c;
        const int o11 = y1c * WDIM + x1c;

        #pragma unroll 1
        for (int cb = 0; cb < NCB; ++cb) {
            const int c0 = cb * CBLK;
            __syncthreads();   // previous step's LDS readers done

            // ---- stage A tile: 64 x 32 f16 (deformable im2col, L2-resident
            //      gathers), packed to a single ds_store_b128 per thread
            {
                const float* bp = inImg + (size_t)(c0 + fc0) * HW;
                h8 av;
                #pragma unroll
                for (int cc = 0; cc < 8; ++cc) {
                    const float* p = bp + (size_t)cc * HW;
                    const float v = w00 * p[o00] + w01 * p[o01]
                                  + w10 * p[o10] + w11 * p[o11];
                    av[cc] = (_Float16)v;
                }
                *(h8*)(As + fm * A_STR + fc0) = av;
            }
            // ---- stage B tile: 256 x 32 f16, row n = tid
            {
                h8* dst = (h8*)(Bs + tid * B_STR);   // first 32 halves of row
                if (USE_WS) {
                    // pre-converted f16, tile-ordered: 4 coalesced b128 loads
                    const h8* src =
                        (const h8*)(bw + ((size_t)(tap * NCB + cb) * COUT + tid) * CBLK);
                    #pragma unroll
                    for (int j = 0; j < 4; ++j) dst[j] = src[j];
                } else {
                    // fallback: strided f32 weights, convert in-flight
                    const float* wp = wgt + ((size_t)tid * CIN + c0) * KTAPS + tap;
                    #pragma unroll
                    for (int j = 0; j < 4; ++j) {
                        h8 bv;
                        #pragma unroll
                        for (int e = 0; e < 8; ++e)
                            bv[e] = (_Float16)wp[(size_t)(j * 8 + e) * KTAPS];
                        dst[j] = bv;
                    }
                }
            }
            __syncthreads();

            // ---- A fragment: lanes 0-15 rows M, K {0..7,16..23}; lanes 16-31 K {8..15,24..31}
            const _Float16* ap = As + (wm * 16 + l16) * A_STR + lhi * 8;
            const h8 a0 = *(const h8*)(ap);
            const h8 a1 = *(const h8*)(ap + 16);
            const h16 a = __builtin_shufflevector(a0, a1,
                0, 1, 2, 3, 4, 5, 6, 7, 8, 9, 10, 11, 12, 13, 14, 15);

            #pragma unroll
            for (int nt = 0; nt < 8; ++nt) {
                // B fragment: lane = N, lanes 0-15 K {0..7,16..23}, lanes 16-31 K {8..15,24..31}
                const _Float16* bpt = Bs + (wn * 128 + nt * 16 + l16) * B_STR + lhi * 8;
                const h8 b0 = *(const h8*)(bpt);
                const h8 b1 = *(const h8*)(bpt + 16);
                const h16 b = __builtin_shufflevector(b0, b1,
                    0, 1, 2, 3, 4, 5, 6, 7, 8, 9, 10, 11, 12, 13, 14, 15);
                acc[nt] = __builtin_amdgcn_wmma_f32_16x16x32_f16(
                    false, a, false, b, (short)0, acc[nt], false, false);
            }
        }
    }

    // ---- epilogue: transpose through LDS (2 passes of 128 Cout) for coalesced NCHW stores
    const int em = tid & 63;             // position within tile
    const int eb = (tid >> 6) * 32;      // 32 couts per thread per pass
    float* const obase = out + (size_t)img * COUT * HW + prem0 + em;

    #pragma unroll 1
    for (int pass = 0; pass < 2; ++pass) {
        __syncthreads();                 // LDS free (last WMMA reads done / prev pass read)
        if (wn == pass) {
            #pragma unroll
            for (int nt = 0; nt < 8; ++nt) {
                const int nl = nt * 16 + l16;            // 0..127
                #pragma unroll
                for (int r = 0; r < 8; ++r) {
                    const int ml = wm * 16 + r + lhi * 8; // 0..63
                    Cs[nl * 65 + ml] = acc[nt][r];
                }
            }
        }
        __syncthreads();
        #pragma unroll
        for (int j = 0; j < 32; ++j) {
            const int nl = eb + j;
            obase[(size_t)(pass * 128 + nl) * HW] = Cs[nl * 65 + em];
        }
    }
}

// ---------------------------------------------------------------- launcher
extern "C" void kernel_launch(void* const* d_in, const int* in_sizes, int n_in,
                              void* d_out, int out_size, void* d_ws, size_t ws_size,
                              hipStream_t stream) {
    (void)in_sizes; (void)n_in; (void)out_size;
    const float* inp = (const float*)d_in[0];
    const float* off = (const float*)d_in[1];
    const float* wgt = (const float*)d_in[2];
    float* out = (float*)d_out;
    const int m_tiles = (NIMG * HW) / TM;   // 73728/64 = 1152 blocks

    if (ws_size >= WS_NEED && d_ws != nullptr) {
        _Float16* bw = (_Float16*)d_ws;
        const int prep_blocks = (KTAPS * NCB * COUT * CBLK) / 256;   // 1152
        prep_weights_f16<<<dim3(prep_blocks), dim3(256), 0, stream>>>(wgt, bw);
        deform_conv_wmma<true><<<dim3(m_tiles), dim3(256), 0, stream>>>(
            inp, off, wgt, bw, out);
    } else {
        deform_conv_wmma<false><<<dim3(m_tiles), dim3(256), 0, stream>>>(
            inp, off, wgt, (const _Float16*)nullptr, out);
    }
}
